// TransformerBlock_15779709845534
// MI455X (gfx1250) — matrix operand order repaired and checked by reading, not compiled, not run on hardware
//
#include <hip/hip_runtime.h>
#include <hip/hip_bf16.h>

// ---------------- problem constants ----------------
static constexpr int Bc   = 2;
static constexpr int Sc   = 512;
static constexpr int Dc   = 768;
static constexpr int Hc   = 12;
static constexpr int DHc  = 64;
static constexpr int EAc  = 4;
static constexpr int EFc  = 8;
static constexpr int DFFc = 3072;
static constexpr int Nc   = Bc * Sc;          // 1024 tokens
static constexpr int CAPc = 160;              // ceil(1.25*1024/8)
static constexpr int CP1  = CAPc + 1;         // 161 (includes dummy slot)

typedef __attribute__((ext_vector_type(16))) __bf16 v16bf;
typedef __attribute__((ext_vector_type(8)))  __bf16 bf16x8;
typedef __attribute__((ext_vector_type(8)))  float  v8f;

// ---------------- LayerNorm (one block per token) ----------------
__global__ __launch_bounds__(256) void ln_kernel(const float* __restrict__ x,
                                                 const float* __restrict__ g,
                                                 const float* __restrict__ b,
                                                 float* __restrict__ out) {
  int row = blockIdx.x;
  const float* xr = x + (size_t)row * Dc;
  __shared__ float red[256];
  int tid = threadIdx.x;
  float s = 0.f;
  for (int d = tid; d < Dc; d += 256) s += xr[d];
  red[tid] = s; __syncthreads();
  for (int st = 128; st > 0; st >>= 1) { if (tid < st) red[tid] += red[tid + st]; __syncthreads(); }
  float mu = red[0] / (float)Dc;
  __syncthreads();
  float v = 0.f;
  for (int d = tid; d < Dc; d += 256) { float t = xr[d] - mu; v += t * t; }
  red[tid] = v; __syncthreads();
  for (int st = 128; st > 0; st >>= 1) { if (tid < st) red[tid] += red[tid + st]; __syncthreads(); }
  float rstd = rsqrtf(red[0] / (float)Dc + 1e-5f);
  for (int d = tid; d < Dc; d += 256)
    out[(size_t)row * Dc + d] = (xr[d] - mu) * rstd * g[d] + b[d];
}

// ---------------- tile staging helpers ----------------
// Stage a ROWSx32 fp32 tile into LDS [row][k] as bf16 (ROWS = 64 or 128).
// TRANS=0: source is row-major rows x K  -> src[row*ld + k]  (contiguous in k)
// TRANS=1: source is k-major            -> src[k*ld + row]   (contiguous in row)
template <int TRANS, int ROWS>
__device__ __forceinline__ void stage_tile(__bf16 (*__restrict__ dst)[40],
                                           const float* __restrict__ src,
                                           int ld, int r0, int k0,
                                           int Rmax, int Kmax, int tid,
                                           bool prefetchNext) {
  #pragma unroll
  for (int rr = 0; rr < ROWS; rr += 64) {
    if (TRANS == 0) {
      int row = (tid >> 2) + rr;     // chunk of 64 rows
      int kq  = (tid & 3) * 8;       // 0,8,16,24
      int gr = r0 + row, gk = k0 + kq;
      if (r0 + rr + 64 <= Rmax && k0 + 32 <= Kmax) {
        const float4* p = (const float4*)(src + (size_t)gr * ld + gk);
        float4 a0 = p[0], a1 = p[1];
        if (prefetchNext) __builtin_prefetch((const char*)(p) + 32 * sizeof(float), 0, 1);
        bf16x8 pk;
        pk[0] = (__bf16)a0.x; pk[1] = (__bf16)a0.y; pk[2] = (__bf16)a0.z; pk[3] = (__bf16)a0.w;
        pk[4] = (__bf16)a1.x; pk[5] = (__bf16)a1.y; pk[6] = (__bf16)a1.z; pk[7] = (__bf16)a1.w;
        *(bf16x8*)&dst[row][kq] = pk;
      } else {
        #pragma unroll
        for (int j = 0; j < 8; ++j) {
          float v = 0.f;
          if (gr < Rmax && gk + j < Kmax) v = src[(size_t)gr * ld + gk + j];
          dst[row][kq + j] = (__bf16)v;
        }
      }
    } else {
      int k  = tid >> 3;             // 0..31
      int rq = (tid & 7) * 8 + rr;   // chunk of 64 rows
      int gk = k0 + k;
      int gr = r0 + rq;
      if (r0 + rr + 64 <= Rmax && k0 + 32 <= Kmax) {
        const float4* p = (const float4*)(src + (size_t)gk * ld + gr);
        float4 a0 = p[0], a1 = p[1];
        if (prefetchNext) __builtin_prefetch((const char*)(src + (size_t)(gk + 32) * ld + gr), 0, 1);
        dst[rq + 0][k] = (__bf16)a0.x; dst[rq + 1][k] = (__bf16)a0.y;
        dst[rq + 2][k] = (__bf16)a0.z; dst[rq + 3][k] = (__bf16)a0.w;
        dst[rq + 4][k] = (__bf16)a1.x; dst[rq + 5][k] = (__bf16)a1.y;
        dst[rq + 6][k] = (__bf16)a1.z; dst[rq + 7][k] = (__bf16)a1.w;
      } else {
        #pragma unroll
        for (int j = 0; j < 8; ++j) {
          float v = 0.f;
          if (gr + j < Rmax && gk < Kmax) v = src[(size_t)gk * ld + gr + j];
          dst[rq + j][k] = (__bf16)v;
        }
      }
    }
  }
}

// ---------------- fragment loaders (ISA 7.12.2 VGPR layouts) ----------------
__device__ __forceinline__ v16bf load_frag_a(const __bf16 (*__restrict__ Xs)[40],
                                             int rbase, int hf, int lm) {
  v16bf f;
  #pragma unroll
  for (int vv = 0; vv < 8; ++vv) {
    // A 16x32 bf16: lanes 0-15 -> K 0..7,16..23 ; lanes 16-31 -> +8
    int ka = ((vv >= 4) ? 16 : 0) + (vv & 3) * 2 + hf * 8;
    f[2 * vv]     = Xs[rbase + lm][ka];
    f[2 * vv + 1] = Xs[rbase + lm][ka + 1];
  }
  return f;
}
__device__ __forceinline__ v16bf load_frag_b(const __bf16 (*__restrict__ Xs)[40],
                                             int rbase, int hf, int lm) {
  v16bf f;
  #pragma unroll
  for (int vv = 0; vv < 8; ++vv) {
    // B 32x16 bf16: lanes 0-15 -> K 0..7,16..23 ; lanes 16-31 -> +8
    int kb = ((vv >= 4) ? 16 : 0) + (vv & 3) * 2 + hf * 8;
    f[2 * vv]     = Xs[rbase + lm][kb];
    f[2 * vv + 1] = Xs[rbase + lm][kb + 1];
  }
  return f;
}

// ---------------- generic bf16-WMMA tiled GEMM ----------------
// C[M,N] = epilogue( alpha * op(A)[M,K] @ op(B)[K,N] + mask )
// Block = 256 threads (8 wave32) in a 4x2 wave grid; block tile 128x64;
// each wave computes a 32x32 patch via 2x2 register blocking (4 WMMAs / K-step).
template <int TA, int TB>
__global__ __launch_bounds__(256) void gemm_wmma_kernel(
    const float* __restrict__ A, const float* __restrict__ B, float* __restrict__ C,
    int M, int N, int Kd, int lda, int ldb, int ldc,
    long long sA1, long long sA2, long long sB1, long long sB2,
    long long sC1, long long sC2, int zdiv,
    float alpha, const float* __restrict__ maskPtr, int maskld, int relu) {
  int z  = blockIdx.z;
  int z1 = z / zdiv;
  int z2 = z - z1 * zdiv;
  A += z1 * sA1 + z2 * sA2;
  B += z1 * sB1 + z2 * sB2;
  C += z1 * sC1 + z2 * sC2;

  int n0 = blockIdx.x * 64;
  int m0 = blockIdx.y * 128;

  __shared__ __bf16 As[128][40];   // [m][k], 80B rows keep 16B vector stores aligned
  __shared__ __bf16 Bs[64][40];    // [n][k]

  int tid  = threadIdx.x;
  int wave = tid >> 5;
  int lane = tid & 31;
  int hf   = lane >> 4;            // half-wave select
  int lm   = lane & 15;
  int wr   = (wave >> 1) * 32;     // wave row origin within block tile
  int wc   = (wave & 1) * 32;      // wave col origin

  v8f zero = {};
  v8f acc[4];                      // [i*2+j] : (i,j) 16x16 subtile of wave's 32x32
  acc[0] = zero; acc[1] = zero; acc[2] = zero; acc[3] = zero;

  for (int k0 = 0; k0 < Kd; k0 += 32) {
    bool pf = (k0 + 32 < Kd);
    stage_tile<TA, 128>(As, A, lda, m0, k0, M, Kd, tid, pf);
    stage_tile<TB, 64>(Bs, B, ldb, n0, k0, N, Kd, tid, pf);
    __syncthreads();

    v16bf fa0 = load_frag_a(As, wr, hf, lm);
    v16bf fa1 = load_frag_a(As, wr + 16, hf, lm);
    v16bf fb0 = load_frag_b(Bs, wc, hf, lm);
    v16bf fb1 = load_frag_b(Bs, wc + 16, hf, lm);

    acc[0] = __builtin_amdgcn_wmma_f32_16x16x32_bf16(false, fa0, false, fb0, (short)0, acc[0], false, false);
    acc[1] = __builtin_amdgcn_wmma_f32_16x16x32_bf16(false, fa0, false, fb1, (short)0, acc[1], false, false);
    acc[2] = __builtin_amdgcn_wmma_f32_16x16x32_bf16(false, fa1, false, fb0, (short)0, acc[2], false, false);
    acc[3] = __builtin_amdgcn_wmma_f32_16x16x32_bf16(false, fa1, false, fb1, (short)0, acc[3], false, false);
    __syncthreads();
  }

  // ---- epilogue: C/D layout -> lane half selects M base, VGPR r = row ----
  #pragma unroll
  for (int i = 0; i < 2; ++i) {
    #pragma unroll
    for (int j = 0; j < 2; ++j) {
      #pragma unroll
      for (int r = 0; r < 8; ++r) {
        int row = m0 + wr + i * 16 + hf * 8 + r;
        int col = n0 + wc + j * 16 + lm;
        if (row < M && col < N) {
          float val = acc[i * 2 + j][r] * alpha;
          if (maskPtr) val += maskPtr[(size_t)row * maskld + col];
          if (relu) val = fmaxf(val, 0.f);
          C[(size_t)row * ldc + col] = val;
        }
      }
    }
  }
}

// ---------------- router: gate = xn @ router_w, argmax over EA per head ----------------
__global__ __launch_bounds__(64) void router_kernel(const float* __restrict__ xn,
                                                    const float* __restrict__ rw,
                                                    int* __restrict__ idx,
                                                    float* __restrict__ head_counts) {
  int n = blockIdx.x;
  __shared__ float xr[Dc];
  __shared__ float gl[Hc * EAc];
  int tid = threadIdx.x;
  for (int d = tid; d < Dc; d += 64) xr[d] = xn[(size_t)n * Dc + d];
  __syncthreads();
  if (tid < Hc * EAc) {
    float a = 0.f;
    for (int d = 0; d < Dc; ++d) a += xr[d] * rw[(size_t)d * (Hc * EAc) + tid];
    gl[tid] = a;
  }
  __syncthreads();
  if (tid < Hc) {
    int best = 0; float bv = gl[tid * EAc];
    for (int e = 1; e < EAc; ++e) { float v = gl[tid * EAc + e]; if (v > bv) { bv = v; best = e; } }
    idx[n * Hc + tid] = best;
    atomicAdd(&head_counts[tid * EAc + best], 1.0f);   // integer-valued -> order-exact
  }
}

// ---------------- per-token expert-routed 64x64 matvec: v_proj ----------------
__global__ __launch_bounds__(64) void vproj_kernel(const float* __restrict__ xn,
                                                   const float* __restrict__ Wv,
                                                   const int* __restrict__ idx,
                                                   float* __restrict__ vproj) {
  int n = blockIdx.x, h = blockIdx.y, d = threadIdx.x;
  int b = n / Sc, s = n % Sc;
  __shared__ float xv[DHc];
  xv[d] = xn[(size_t)n * Dc + h * DHc + d];
  __syncthreads();
  int e = idx[n * Hc + h];
  const float* W = Wv + (((size_t)h * EAc + e) * DHc) * DHc;
  float acc = 0.f;
  for (int k = 0; k < DHc; ++k) acc += xv[k] * W[k * DHc + d];
  vproj[((((size_t)h * Bc + b) * Sc) + s) * DHc + d] = acc;
}

// ---------------- o_proj + residual ----------------
__global__ __launch_bounds__(64) void oproj_kernel(const float* __restrict__ attn,
                                                   const float* __restrict__ Wo,
                                                   const int* __restrict__ idx,
                                                   const float* __restrict__ x,
                                                   float* __restrict__ x1) {
  int n = blockIdx.x, h = blockIdx.y, d = threadIdx.x;
  int b = n / Sc, s = n % Sc;
  __shared__ float av[DHc];
  av[d] = attn[((((size_t)h * Bc + b) * Sc) + s) * DHc + d];
  __syncthreads();
  int e = idx[n * Hc + h];
  const float* W = Wo + ((size_t)h * EAc + e) * DHc * DHc;
  float acc = 0.f;
  for (int k = 0; k < DHc; ++k) acc += av[k] * W[k * DHc + d];
  size_t o = (size_t)n * Dc + h * DHc + d;
  x1[o] = x[o] + acc;
}

// ---------------- row softmax (one block per score row of length 512) ----------------
__global__ __launch_bounds__(256) void softmax_kernel(float* __restrict__ scores) {
  int row = blockIdx.x;
  float* r = scores + (size_t)row * Sc;
  __shared__ float red[256];
  int tid = threadIdx.x;
  float m = -3.4e38f;
  for (int t = tid; t < Sc; t += 256) m = fmaxf(m, r[t]);
  red[tid] = m; __syncthreads();
  for (int st = 128; st > 0; st >>= 1) { if (tid < st) red[tid] = fmaxf(red[tid], red[tid + st]); __syncthreads(); }
  m = red[0]; __syncthreads();
  float s = 0.f;
  for (int t = tid; t < Sc; t += 256) { float e = expf(r[t] - m); r[t] = e; s += e; }
  red[tid] = s; __syncthreads();
  for (int st = 128; st > 0; st >>= 1) { if (tid < st) red[tid] += red[tid + st]; __syncthreads(); }
  float inv = 1.f / red[0];
  for (int t = tid; t < Sc; t += 256) r[t] *= inv;
}

// ---------------- MoE gate + top-2 softmax ----------------
__global__ __launch_bounds__(64) void gate_topk_kernel(const float* __restrict__ xn,
                                                       const float* __restrict__ gw,
                                                       int* __restrict__ tki,
                                                       float* __restrict__ tkp) {
  int n = blockIdx.x;
  __shared__ float xr[Dc];
  __shared__ float gl[EFc];
  int tid = threadIdx.x;
  for (int d = tid; d < Dc; d += 64) xr[d] = xn[(size_t)n * Dc + d];
  __syncthreads();
  if (tid < EFc) {
    float a = 0.f;
    for (int d = 0; d < Dc; ++d) a += xr[d] * gw[(size_t)d * EFc + tid];
    gl[tid] = a;
  }
  __syncthreads();
  if (tid == 0) {
    int i0 = 0; float v0 = gl[0];
    for (int e = 1; e < EFc; ++e) if (gl[e] > v0) { v0 = gl[e]; i0 = e; }
    int i1 = (i0 == 0) ? 1 : 0; float v1 = gl[i1];
    for (int e = 0; e < EFc; ++e) { if (e == i0) continue; if (gl[e] > v1) { v1 = gl[e]; i1 = e; } }
    float mx = fmaxf(v0, v1);
    float e0 = expf(v0 - mx), e1 = expf(v1 - mx);
    float ss = e0 + e1;
    tki[n * 2] = i0; tki[n * 2 + 1] = i1;
    tkp[n * 2] = e0 / ss; tkp[n * 2 + 1] = e1 / ss;
  }
}

// ---------------- serial capacity scan (token-major cumulative positions) ----------------
__global__ void scan_kernel(const int* __restrict__ tki, const float* __restrict__ tkp,
                            int* __restrict__ pos_c, float* __restrict__ w,
                            float* __restrict__ counts, float* __restrict__ importance) {
  if (threadIdx.x != 0 || blockIdx.x != 0) return;
  int cnt[EFc];
  for (int e = 0; e < EFc; ++e) cnt[e] = 0;
  for (int n = 0; n < Nc; ++n) {
    float p[2]; int pc[2];
    for (int k = 0; k < 2; ++k) {
      int e = tki[n * 2 + k];
      int pos = cnt[e]++;
      bool keep = pos < CAPc;
      pc[k] = keep ? pos : CAPc;           // dropped -> dummy slot
      p[k] = keep ? tkp[n * 2 + k] : 0.f;
    }
    float ssum = p[0] + p[1] + 1e-9f;
    for (int k = 0; k < 2; ++k) {
      float wk = p[k] / ssum;
      w[n * 2 + k] = wk;
      pos_c[n * 2 + k] = pc[k];
      int e = tki[n * 2 + k];
      if (wk > 0.f) counts[e] += 1.f;
      importance[e] += wk;
    }
  }
}

// ---------------- dispatch tokens into expert capacity buffers ----------------
__global__ __launch_bounds__(256) void dispatch_kernel(const float* __restrict__ xn2,
                                                       const int* __restrict__ tki,
                                                       const int* __restrict__ pos_c,
                                                       float* __restrict__ buf) {
  int n = blockIdx.x, tid = threadIdx.x;
  const float* src = xn2 + (size_t)n * Dc;
  for (int k = 0; k < 2; ++k) {
    int e = tki[n * 2 + k];
    int p = pos_c[n * 2 + k];
    float* dst = buf + ((size_t)e * CP1 + p) * Dc;
    for (int d = tid; d < Dc; d += 256) dst[d] = src[d];
  }
}

// ---------------- combine: y = x1 + sum_k w_k * ybuf[e_k, pos_k] ----------------
__global__ __launch_bounds__(256) void combine_kernel(const float* __restrict__ x1,
                                                      const float* __restrict__ ybuf,
                                                      const int* __restrict__ tki,
                                                      const int* __restrict__ pos_c,
                                                      const float* __restrict__ w,
                                                      float* __restrict__ out) {
  int n = blockIdx.x, tid = threadIdx.x;
  int e0 = tki[n * 2], e1 = tki[n * 2 + 1];
  int p0 = pos_c[n * 2], p1 = pos_c[n * 2 + 1];
  float w0 = w[n * 2], w1 = w[n * 2 + 1];
  const float* y0 = ybuf + ((size_t)e0 * CP1 + p0) * Dc;
  const float* y1 = ybuf + ((size_t)e1 * CP1 + p1) * Dc;
  for (int d = tid; d < Dc; d += 256)
    out[(size_t)n * Dc + d] = x1[(size_t)n * Dc + d] + w0 * y0[d] + w1 * y1[d];
}

// ---------------- aux losses ----------------
__global__ void finalize_kernel(const float* __restrict__ head_counts,
                                const float* __restrict__ counts,
                                const float* __restrict__ importance,
                                float* __restrict__ out_aux) {
  if (threadIdx.x != 0 || blockIdx.x != 0) return;
  float ema[Hc * EAc]; float ssum = 0.f;
  for (int i = 0; i < Hc * EAc; ++i) { ema[i] = 0.01f * head_counts[i] / (float)Nc; ssum += ema[i]; }
  float inv = 1.f / (ssum + 1e-9f);
  float aux1 = 0.f;
  for (int i = 0; i < Hc * EAc; ++i) { float pr = ema[i] * inv; aux1 += pr * pr; }
  aux1 *= (float)(EAc * Hc);
  float cs = 0.f, is_ = 0.f;
  for (int e = 0; e < EFc; ++e) { cs += counts[e]; is_ += importance[e]; }
  float aux2 = 0.f;
  for (int e = 0; e < EFc; ++e) aux2 += (counts[e] / (cs + 1e-9f)) * (importance[e] / (is_ + 1e-9f));
  aux2 *= (float)EFc;
  *out_aux = aux1 + aux2;
}

// ==================== launcher ====================
extern "C" void kernel_launch(void* const* d_in, const int* in_sizes, int n_in,
                              void* d_out, int out_size, void* d_ws, size_t ws_size,
                              hipStream_t stream) {
  (void)in_sizes; (void)n_in; (void)out_size; (void)ws_size;
  const float* x        = (const float*)d_in[0];
  const float* mask     = (const float*)d_in[1];
  const float* ln1_g    = (const float*)d_in[2];
  const float* ln1_b    = (const float*)d_in[3];
  const float* ln2_g    = (const float*)d_in[4];
  const float* ln2_b    = (const float*)d_in[5];
  const float* W_q      = (const float*)d_in[6];
  const float* W_k      = (const float*)d_in[7];
  const float* W_v      = (const float*)d_in[8];
  const float* W_o      = (const float*)d_in[9];
  const float* router_w = (const float*)d_in[10];
  const float* gate_w   = (const float*)d_in[11];
  const float* W1       = (const float*)d_in[12];
  const float* W2       = (const float*)d_in[13];
  float* out = (float*)d_out;

  // ---- workspace partition (floats) ----
  char* cur = (char*)d_ws;
  auto allocf = [&](size_t nelem) { float* p = (float*)cur; cur += nelem * sizeof(float); return p; };
  auto alloci = [&](size_t nelem) { int* p = (int*)cur; cur += nelem * sizeof(int); return p; };
  float* xn1     = allocf((size_t)Nc * Dc);                 // LN1(x)
  float* Qm      = allocf((size_t)Nc * Dc);
  float* Km      = allocf((size_t)Nc * Dc);
  float* vproj   = allocf((size_t)Hc * Nc * DHc);
  float* scores  = allocf((size_t)Bc * Hc * Sc * Sc);       // 25 MB
  float* attn    = allocf((size_t)Hc * Nc * DHc);
  float* x1      = allocf((size_t)Nc * Dc);
  float* xn2     = allocf((size_t)Nc * Dc);
  float* buf     = allocf((size_t)EFc * CP1 * Dc);
  float* hmid    = allocf((size_t)EFc * CP1 * DFFc);
  float* ybuf    = allocf((size_t)EFc * CP1 * Dc);
  int*   idx     = alloci(Nc * Hc);
  int*   tki     = alloci(Nc * 2);
  int*   pos_c   = alloci(Nc * 2);
  float* tkp     = allocf(Nc * 2);
  float* wdisp   = allocf(Nc * 2);
  float* hcounts = allocf(64);
  float* counts  = allocf(16);
  float* importn = allocf(16);

  // zero the buffers that need it (capture-safe)
  hipMemsetAsync(hcounts, 0, 64 * sizeof(float), stream);
  hipMemsetAsync(counts, 0, 16 * sizeof(float), stream);
  hipMemsetAsync(importn, 0, 16 * sizeof(float), stream);
  hipMemsetAsync(buf, 0, (size_t)EFc * CP1 * Dc * sizeof(float), stream);

  // 1) LN1
  ln_kernel<<<Nc, 256, 0, stream>>>(x, ln1_g, ln1_b, xn1);

  // 2) Q = xn1 @ W_q ; K = xn1 @ W_k   (1024x768x768, bf16 WMMA)
  {
    dim3 grid(Dc / 64, Nc / 128, 1);
    gemm_wmma_kernel<0, 0><<<grid, 256, 0, stream>>>(xn1, W_q, Qm, Nc, Dc, Dc, Dc, Dc, Dc,
        0, 0, 0, 0, 0, 0, 1, 1.0f, nullptr, 0, 0);
    gemm_wmma_kernel<0, 0><<<grid, 256, 0, stream>>>(xn1, W_k, Km, Nc, Dc, Dc, Dc, Dc, Dc,
        0, 0, 0, 0, 0, 0, 1, 1.0f, nullptr, 0, 0);
  }

  // 3) router argmax + head histogram; 4) routed v_proj
  router_kernel<<<Nc, 64, 0, stream>>>(xn1, router_w, idx, hcounts);
  {
    dim3 grid(Nc, Hc, 1);
    vproj_kernel<<<grid, 64, 0, stream>>>(xn1, W_v, idx, vproj);
  }

  // 5) scores = Q Kᵀ / 8 + mask  per (b,h): z = b*H + h ; Kᵀ access is contiguous-in-n (TB=1)
  {
    dim3 grid(Sc / 64, Sc / 128, Bc * Hc);
    gemm_wmma_kernel<0, 1><<<grid, 256, 0, stream>>>(
        Qm, Km, scores, Sc, Sc, DHc, Dc, Dc, Sc,
        /*sA1(b)=*/(long long)Sc * Dc, /*sA2(h)=*/DHc,
        /*sB1(b)=*/(long long)Sc * Dc, /*sB2(h)=*/DHc,
        /*sC1(b)=*/(long long)Hc * Sc * Sc, /*sC2(h)=*/(long long)Sc * Sc,
        /*zdiv=*/Hc, /*alpha=*/0.125f, mask, Sc, 0);
  }

  // 6) row softmax
  softmax_kernel<<<Bc * Hc * Sc, 256, 0, stream>>>(scores);

  // 7) attn_out = Pᵀ @ V  per (h,b): z = h*B + b ; P accessed transposed (TA=1)
  {
    dim3 grid(DHc / 64, Sc / 128, Hc * Bc);
    gemm_wmma_kernel<1, 0><<<grid, 256, 0, stream>>>(
        scores, vproj, attn, Sc, DHc, Sc, Sc, DHc, DHc,
        /*sA1(h)=*/(long long)Sc * Sc, /*sA2(b)=*/(long long)Hc * Sc * Sc,
        /*sB1(h)=*/(long long)Bc * Sc * DHc, /*sB2(b)=*/(long long)Sc * DHc,
        /*sC1(h)=*/(long long)Bc * Sc * DHc, /*sC2(b)=*/(long long)Sc * DHc,
        /*zdiv=*/Bc, 1.0f, nullptr, 0, 0);
  }

  // 8) o_proj + residual -> x1
  {
    dim3 grid(Nc, Hc, 1);
    oproj_kernel<<<grid, 64, 0, stream>>>(attn, W_o, idx, x, x1);
  }

  // 9) LN2, gate/top-2, capacity scan, dispatch
  ln_kernel<<<Nc, 256, 0, stream>>>(x1, ln2_g, ln2_b, xn2);
  gate_topk_kernel<<<Nc, 64, 0, stream>>>(xn2, gate_w, tki, tkp);
  scan_kernel<<<1, 32, 0, stream>>>(tki, tkp, pos_c, wdisp, counts, importn);
  dispatch_kernel<<<Nc, 256, 0, stream>>>(xn2, tki, pos_c, buf);

  // 10) expert FFN GEMM1: hmid = relu(buf @ W1)  per expert (161x3072x768)
  {
    dim3 grid(DFFc / 64, (CP1 + 127) / 128, EFc);
    gemm_wmma_kernel<0, 0><<<grid, 256, 0, stream>>>(
        buf, W1, hmid, CP1, DFFc, Dc, Dc, DFFc, DFFc,
        (long long)CP1 * Dc, 0, (long long)Dc * DFFc, 0, (long long)CP1 * DFFc, 0,
        1, 1.0f, nullptr, 0, /*relu=*/1);
  }
  // 11) expert FFN GEMM2: ybuf = hmid @ W2  per expert (161x768x3072)
  {
    dim3 grid(Dc / 64, (CP1 + 127) / 128, EFc);
    gemm_wmma_kernel<0, 0><<<grid, 256, 0, stream>>>(
        hmid, W2, ybuf, CP1, Dc, DFFc, DFFc, Dc, Dc,
        (long long)CP1 * DFFc, 0, (long long)DFFc * Dc, 0, (long long)CP1 * Dc, 0,
        1, 1.0f, nullptr, 0, 0);
  }

  // 12) combine + residual -> y ; 13) aux scalar
  combine_kernel<<<Nc, 256, 0, stream>>>(x1, ybuf, tki, pos_c, wdisp, out);
  finalize_kernel<<<1, 32, 0, stream>>>(hcounts, counts, importn, out + (size_t)Nc * Dc);
}
